// MPnet_82935818486482
// MI455X (gfx1250) — matrix operand order repaired and checked by reading, not compiled, not run on hardware
//
#include <hip/hip_runtime.h>
#include <hip/hip_bf16.h>
#include <cstdint>

// ---------------------------------------------------------------------------
// MI455X / gfx1250, wave32.
//  - bf16 WMMA (v_wmma_f32_16x16x32_bf16) GEMM core
//  - Tensor Data Mover (tensor_load_to_lds + s_wait_tensorcnt) for LDS staging
// ---------------------------------------------------------------------------

typedef __attribute__((ext_vector_type(16))) __bf16 v16bf;
typedef __attribute__((ext_vector_type(8)))  float  v8f;
typedef __attribute__((ext_vector_type(4)))  unsigned int v4u;
typedef __attribute__((ext_vector_type(8)))  int    v8i;
typedef __attribute__((ext_vector_type(4)))  int    v4i;

#define TPB 256

__device__ __forceinline__ unsigned short f2bf(float f) {
  unsigned u = __float_as_uint(f);
  u += 0x7FFFu + ((u >> 16) & 1u);          // round-to-nearest-even
  return (unsigned short)(u >> 16);
}
__device__ __forceinline__ __bf16 bfbits(unsigned short s) {
  return __builtin_bit_cast(__bf16, s);
}
__device__ __forceinline__ float sigf(float x) { return 1.0f / (1.0f + __expf(-x)); }

// ---------------------------------------------------------------------------
// TDM: async 2D tile (2-byte elements) global -> LDS.
//   tile_d0   : contiguous extent (elements)
//   tile_d1   : number of rows
//   tensor_d0/1: remaining tensor extents (OOB reads return zero -> edge fill)
//   stride0   : row stride in elements
//   pad_code  : D# pad_interval code (3 = pad after 16 dwords, 4 = after 32)
//               with pad_amount = 1 dword (matches LDS row pads below)
// ---------------------------------------------------------------------------
__device__ __forceinline__ void tdm_load_2d(unsigned lds_addr, const void* gptr,
                                            int tile_d0, int tile_d1,
                                            long long tensor_d0, long long tensor_d1,
                                            long long stride0, int pad_code) {
  unsigned long long ga = (unsigned long long)(uintptr_t)gptr;
  unsigned d0 = (unsigned)(tensor_d0 < 0 ? 0 : tensor_d0);
  unsigned d1 = (unsigned)(tensor_d1 < 0 ? 0 : tensor_d1);
  unsigned long long st0 = (unsigned long long)stride0;

  v4u g0;
  g0[0] = 1u;                                           // count=1, user mode
  g0[1] = lds_addr;                                     // lds_addr
  g0[2] = (unsigned)(ga & 0xFFFFFFFFu);                 // global_addr lo
  g0[3] = (unsigned)((ga >> 32) & 0x01FFFFFFu)          // global_addr hi
        | 0x80000000u;                                  // type=2 ("image")

  v8i g1;
  g1[0] = (int)((1u << 16)                              // data_size = 2 bytes
              | (1u << 20)                              // pad_enable
              | ((unsigned)pad_code << 22));            // pad_interval; amount=1 dword
  g1[1] = (int)((d0 & 0xFFFFu) << 16);                  // tensor_dim0[15:0]
  g1[2] = (int)((d0 >> 16) | ((d1 & 0xFFFFu) << 16));   // dim0[31:16] | dim1[15:0]
  g1[3] = (int)((d1 >> 16) | ((unsigned)tile_d0 << 16));// dim1[31:16] | tile_dim0
  g1[4] = (int)((unsigned)tile_d1 & 0xFFFFu);           // tile_dim1; tile_dim2=0
  g1[5] = (int)(unsigned)(st0 & 0xFFFFFFFFu);           // dim0_stride lo
  g1[6] = (int)(unsigned)((st0 >> 32) & 0xFFFFu);       // dim0_stride hi
  g1[7] = 0;

  v4i gz = {0, 0, 0, 0};
#if __clang_major__ >= 23
  v8i gz8 = {0, 0, 0, 0, 0, 0, 0, 0};
  __builtin_amdgcn_tensor_load_to_lds(g0, g1, gz, gz, gz8, 0);
#else
  __builtin_amdgcn_tensor_load_to_lds(g0, g1, gz, gz, 0);
#endif
}

// ---------------------------------------------------------------------------
// Templated batched GEMM:  C[M,N] = A[M,K] (bf16) * B (bf16) + bias
//   BMODE 0: B stored [N,K] row-major (C = A @ W^T)
//   BMODE 1: B stored [K,N] row-major (C = A @ W)
//   BMODE 2: conv: A = kk-planes (3,M,K=Cin) of permuted weights, lda=Cin;
//            B[k,n] tap kk = Bp[k*ldb + kk*dil + n] (ldb = Lin)
// Block: 256 threads = 8 waves; tile 32(M) x 64(N); K staged in 32-chunks.
// All staging via TDM (wave 0 issues, s_wait_tensorcnt, barrier).
// ---------------------------------------------------------------------------
template <int BMODE>
__global__ __launch_bounds__(TPB)
void gemm_bf16_wmma_t(const unsigned short* __restrict__ A, long long lda,
                      const unsigned short* __restrict__ Bm, long long ldb, long long sB,
                      int dil,
                      const float* __restrict__ bias,
                      float* __restrict__ Cf, unsigned short* __restrict__ Cb,
                      long long ldc, long long sC,
                      int M, int N, int K)
{
  __shared__ unsigned short As[32][34];        // [m][k], row = 16+1 dwords
  __shared__ unsigned short BsRaw[64 * 34];    // union of the two B layouts
  auto BsNK = reinterpret_cast<unsigned short(*)[34]>(&BsRaw[0]);  // [n][k]
  auto BsKN = reinterpret_cast<unsigned short(*)[66]>(&BsRaw[0]);  // [k][n], row = 32+1 dwords

  const int tid = threadIdx.x;
  const long long z = blockIdx.z;
  Bm += z * sB;
  Cf += z * sC;
  if (Cb) Cb += z * sC;

  const int m0 = blockIdx.y * 32;
  const int n0 = blockIdx.x * 64;
  const int wave = tid >> 5;
  const int lane = tid & 31;
  const int wm = wave >> 2;   // 0..1
  const int wn = wave & 3;    // 0..3
  const int l16 = lane & 15;
  const int hs  = lane >> 4;

  const unsigned ldsA = (unsigned)(uintptr_t)(&As[0][0]);
  const unsigned ldsB = (unsigned)(uintptr_t)(&BsRaw[0]);

  const int TAPS = (BMODE == 2) ? 3 : 1;
  const long long planeA = (long long)M * K;   // kk-plane stride (conv weights)

  v8f acc = {};

  for (int kb = 0; kb < K; kb += 32) {
    for (int kk = 0; kk < TAPS; ++kk) {
      if (tid < 32) {  // wave 0 drives the Tensor Data Mover
        const unsigned short* Ab = A + (BMODE == 2 ? (long long)kk * planeA : 0)
                                     + (long long)m0 * lda + kb;
        tdm_load_2d(ldsA, Ab, 32, 32, (long long)K - kb, (long long)M - m0, lda, 3);
        if (BMODE == 0) {
          tdm_load_2d(ldsB, Bm + (long long)n0 * ldb + kb, 32, 64,
                      (long long)K - kb, (long long)N - n0, ldb, 3);
        } else {
          long long boff = (long long)kb * ldb + n0
                         + (BMODE == 2 ? (long long)kk * dil : 0);
          long long rem0 = (BMODE == 2) ? (ldb - (long long)kk * dil - n0)
                                        : ((long long)N - n0);
          tdm_load_2d(ldsB, Bm + boff, 64, 32, rem0, (long long)K - kb, ldb, 4);
        }
        __builtin_amdgcn_s_wait_tensorcnt((short)0);
      }
      __syncthreads();

      // Fragments (wave32): A and B both hold k = hs*16 + i (contiguous k order).
      v16bf af, bfr;
      const int ar = wm * 16 + l16;
      const int bc = wn * 16 + l16;
#pragma unroll
      for (int i = 0; i < 8; ++i) {
        af[i]     = bfbits(As[ar][hs * 16 + i]);      // K = hs*16 + i
        af[i + 8] = bfbits(As[ar][hs * 16 + 8 + i]);  // K = hs*16 + 8 + i
      }
#pragma unroll
      for (int i = 0; i < 16; ++i) {
        bfr[i] = (BMODE == 0) ? bfbits(BsNK[bc][hs * 16 + i])
                              : bfbits(BsKN[hs * 16 + i][bc]);
      }
      acc = __builtin_amdgcn_wmma_f32_16x16x32_bf16(
          /*neg_a=*/false, af, /*neg_b=*/false, bfr,
          /*c_mod=*/(short)0, acc, /*reuse_a=*/false, /*reuse_b=*/false);
      __syncthreads();
    }
  }

  // C/D layout: acc[r] = C(m0+wm*16 + hs*8 + r, n0+wn*16 + l16)
  const int gn = n0 + wn * 16 + l16;
  const float bv = (bias && gn < N) ? bias[gn] : 0.0f;
#pragma unroll
  for (int r = 0; r < 8; ++r) {
    int gm = m0 + wm * 16 + hs * 8 + r;
    if (gm < M && gn < N) {
      float v = acc[r] + bv;
      Cf[(long long)gm * ldc + gn] = v;
      if (Cb) Cb[(long long)gm * ldc + gn] = f2bf(v);
    }
  }
}

// ---------------------------------------------------------------------------
// Pointwise / gather kernels
// ---------------------------------------------------------------------------
__global__ void embed_bf16_knl(const int* __restrict__ tok, const float* __restrict__ emb,
                               unsigned short* __restrict__ out, long long rows, int E) {
  long long i = (long long)blockIdx.x * blockDim.x + threadIdx.x;
  if (i >= rows * (long long)E) return;
  long long r = i / E; int e = (int)(i % E);
  out[i] = f2bf(emb[(long long)tok[r] * E + e]);
}

__global__ void f32_to_bf16_knl(const float* __restrict__ in, unsigned short* __restrict__ out,
                                long long n) {
  long long i = (long long)blockIdx.x * blockDim.x + threadIdx.x;
  if (i < n) out[i] = f2bf(in[i]);
}

// conv weight (Cout,Cin,3) f32 -> bf16 permuted planes (kk,Cout,Cin)
__global__ void conv_w_perm_knl(const float* __restrict__ src, unsigned short* __restrict__ dst,
                                int Cout, int Cin) {
  long long i = (long long)blockIdx.x * blockDim.x + threadIdx.x;
  long long tot = (long long)Cout * Cin * 3;
  if (i >= tot) return;
  int kk = (int)(i % 3);
  long long t = i / 3;
  int ci = (int)(t % Cin);
  int co = (int)(t / Cin);
  dst[(long long)kk * Cout * Cin + (long long)co * Cin + ci] = f2bf(src[i]);
}

// One GRU gate step (H=512 hard-wired). hsF/hsB have row stride T*1024.
__global__ void gru_point_knl(const float* __restrict__ xg, const float* __restrict__ gh,
                              float* __restrict__ hf, unsigned short* __restrict__ hb,
                              float* __restrict__ hsF, unsigned short* __restrict__ hsB,
                              int Bv, int T, int tx, int tout, int coloff) {
  int i = blockIdx.x * blockDim.x + threadIdx.x;
  if (i >= Bv * 512) return;
  int bv = i >> 9; int j = i & 511;
  const float* xr = xg + ((long long)bv * T + tx) * 1536;
  const float* gr = gh + (long long)bv * 1536;
  float gi_r = xr[j], gi_z = xr[512 + j], gi_n = xr[1024 + j];
  float gh_r = gr[j], gh_z = gr[512 + j], gh_n = gr[1024 + j];
  float h  = hf[i];
  float r  = sigf(gi_r + gh_r);
  float zz = sigf(gi_z + gh_z);
  float n  = tanhf(gi_n + r * gh_n);
  float hn = (1.0f - zz) * n + zz * h;
  hf[i] = hn; hb[i] = f2bf(hn);
  long long o = ((long long)bv * T + tout) * 1024 + coloff + j;
  hsF[o] = hn; hsB[o] = f2bf(hn);
}

__global__ void bh_knl(const float* __restrict__ sent, const float* __restrict__ attn_b,
                       float* __restrict__ bh, int BS, int HH) {
  int i = blockIdx.x * blockDim.x + threadIdx.x;
  if (i >= BS) return;
  const float* row = sent + (long long)i * HH;
  float s = 0.0f;
  for (int h = 0; h < HH; ++h) s += row[h] * attn_b[h];
  bh[i] = s;
}

// Attention read for choice 0 only (only Cs[0] reaches the output).
__global__ void attn_knl(const float* __restrict__ Wh, const float* __restrict__ bh,
                         const float* __restrict__ cv, const float* __restrict__ sent,
                         float* __restrict__ pars, int Sn, int HH) {
  int b = blockIdx.x; int tid = threadIdx.x;
  __shared__ float cvl[1024];
  __shared__ float sc[96];
  __shared__ float inv_sum;
  for (int h = tid; h < HH; h += blockDim.x) cvl[h] = cv[(long long)(b * 4) * HH + h];
  __syncthreads();
  int wv = tid >> 5, lane = tid & 31;
  for (int s = wv; s < Sn; s += 8) {
    const float* wr = Wh + ((long long)b * Sn + s) * HH;
    float v = 0.0f;
    for (int h = lane; h < HH; h += 32) v += wr[h] * cvl[h];
    for (int o = 16; o > 0; o >>= 1) v += __shfl_xor(v, o, 32);
    if (lane == 0) sc[s] = v + bh[b * Sn + s];
  }
  __syncthreads();
  if (tid == 0) {
    float mx = -3.4e38f;
    for (int s = 0; s < Sn; ++s) mx = fmaxf(mx, sc[s]);
    float sum = 0.0f;
    for (int s = 0; s < Sn; ++s) { sc[s] = __expf(sc[s] - mx); sum += sc[s]; }
    inv_sum = 1.0f / sum;
  }
  __syncthreads();
  for (int h = tid; h < HH; h += blockDim.x) {
    float a = 0.0f;
    for (int s = 0; s < Sn; ++s) a += sc[s] * sent[((long long)b * Sn + s) * HH + h];
    pars[(long long)b * HH + h] = a * inv_sum;
  }
}

// bias (+ optional BN) (+ optional ReLU); writes f32 in place + optional bf16.
__global__ void conv_post_knl(float* __restrict__ y, unsigned short* __restrict__ yb,
                              const float* __restrict__ bias,
                              const float* __restrict__ g, const float* __restrict__ bb,
                              const float* __restrict__ mean, const float* __restrict__ var,
                              int relu, int Cc, long long L, long long total) {
  long long i = (long long)blockIdx.x * blockDim.x + threadIdx.x;
  if (i >= total) return;
  int co = (int)((i / L) % Cc);
  float v = y[i] + bias[co];
  if (g) v = (v - mean[co]) * (g[co] * rsqrtf(var[co] + 1e-5f)) + bb[co];
  if (relu) v = fmaxf(v, 0.0f);
  y[i] = v;
  if (yb) yb[i] = f2bf(v);
}

__global__ void maxpool_knl(const float* __restrict__ x, float* __restrict__ Pf,
                            unsigned short* __restrict__ Pb, int rows, int L) {
  int i = blockIdx.x * blockDim.x + threadIdx.x;
  if (i >= rows) return;
  const float* r = x + (long long)i * L;
  float m = -3.4e38f;
  for (int l = 0; l < L; ++l) m = fmaxf(m, r[l]);
  Pf[i] = m; Pb[i] = f2bf(m);
}

__global__ void cs0_knl(const float* __restrict__ cv, const float* __restrict__ pars,
                        float* __restrict__ cs0, unsigned short* __restrict__ cs0b, int Bn) {
  int i = blockIdx.x * blockDim.x + threadIdx.x;
  if (i >= Bn * 2048) return;
  int b = i >> 11; int j = i & 2047;
  float v = (j < 1024) ? cv[(long long)(b * 4) * 1024 + j] : pars[(long long)b * 1024 + (j - 1024)];
  cs0[i] = v; cs0b[i] = f2bf(v);
}

__global__ void final_knl(const float* __restrict__ g1, const float* __restrict__ g2,
                          const float* __restrict__ gate_b, const float* __restrict__ cs0,
                          const float* __restrict__ wpo, const float* __restrict__ out_b,
                          float* __restrict__ out) {
  int b = blockIdx.x; int tid = threadIdx.x;
  __shared__ float red[TPB];
  float s = 0.0f;
  for (int j = tid; j < 2048; j += TPB) {
    long long o = (long long)b * 2048 + j;
    float gv = g1[o] + g2[o] + gate_b[j];
    float cd = cs0[o] * sigf(gv);
    s += cd * (wpo[o] + out_b[j]);
  }
  red[tid] = s; __syncthreads();
  for (int o = TPB / 2; o > 0; o >>= 1) {
    if (tid < o) red[tid] += red[tid + o];
    __syncthreads();
  }
  if (tid == 0) {
    float o1 = red[0];
    out[b * 4 + 0] = o1; out[b * 4 + 1] = o1; out[b * 4 + 2] = o1; out[b * 4 + 3] = o1;
  }
}

// ---------------------------------------------------------------------------
// Host driver
// ---------------------------------------------------------------------------
extern "C" void kernel_launch(void* const* d_in, const int* in_sizes, int n_in,
                              void* d_out, int out_size, void* d_ws, size_t ws_size,
                              hipStream_t stream) {
  (void)in_sizes; (void)n_in; (void)out_size; (void)ws_size;

  const int Bn = 128, Sn = 80, En = 300, Hn = 512, Cn = 6, HHn = 1024, G3 = 1536;
  const int BS  = Bn * Sn;                    // 10240 sentence tokens
  const int BCC = Bn * 4;                     // 512 choice rows
  const long long BCT = (long long)BCC * Cn;  // 3072 choice tokens

  const int*   tok_s   = (const int*)d_in[0];
  const int*   tok_c   = (const int*)d_in[1];
  const float* emb     = (const float*)d_in[2];
  const float* sgf_wih = (const float*)d_in[3];
  const float* sgf_whh = (const float*)d_in[4];
  const float* sgf_bih = (const float*)d_in[5];
  const float* sgf_bhh = (const float*)d_in[6];
  const float* sgb_wih = (const float*)d_in[7];
  const float* sgb_whh = (const float*)d_in[8];
  const float* sgb_bih = (const float*)d_in[9];
  const float* sgb_bhh = (const float*)d_in[10];
  const float* cgf_wih = (const float*)d_in[11];
  const float* cgf_whh = (const float*)d_in[12];
  const float* cgf_bih = (const float*)d_in[13];
  const float* cgf_bhh = (const float*)d_in[14];
  const float* cgb_wih = (const float*)d_in[15];
  const float* cgb_whh = (const float*)d_in[16];
  const float* cgb_bih = (const float*)d_in[17];
  const float* cgb_bhh = (const float*)d_in[18];
  const float* lin_w   = (const float*)d_in[19];
  const float* lin_b   = (const float*)d_in[20];
  const float* attn_w  = (const float*)d_in[21];
  const float* attn_b  = (const float*)d_in[22];
  const float* c11_w   = (const float*)d_in[23];
  const float* c11_b   = (const float*)d_in[24];
  const float* c13_w   = (const float*)d_in[25];
  const float* c13_b   = (const float*)d_in[26];
  const float* c21_w   = (const float*)d_in[27];
  const float* c21_b   = (const float*)d_in[28];
  const float* c23_w   = (const float*)d_in[29];
  const float* c23_b   = (const float*)d_in[30];
  const float* bn1_g   = (const float*)d_in[31];
  const float* bn1_bb  = (const float*)d_in[32];
  const float* bn1_m   = (const float*)d_in[33];
  const float* bn1_v   = (const float*)d_in[34];
  const float* bn2_g   = (const float*)d_in[35];
  const float* bn2_bb  = (const float*)d_in[36];
  const float* bn2_m   = (const float*)d_in[37];
  const float* bn2_v   = (const float*)d_in[38];
  const float* gate_wp = (const float*)d_in[39];
  const float* gate_wc = (const float*)d_in[40];
  const float* gate_b  = (const float*)d_in[41];
  const float* out_wp  = (const float*)d_in[42];
  const float* out_b   = (const float*)d_in[43];

  // ---- workspace bump allocator -------------------------------------------
  char* wsp = (char*)d_ws;
  size_t off = 0;
  auto alloc = [&](size_t bytes) -> void* {
    void* p = wsp + off;
    off = (off + bytes + 255) & ~(size_t)255;
    return p;
  };
  auto aF = [&](long long n) { return (float*)alloc((size_t)n * 4); };
  auto aU = [&](long long n) { return (unsigned short*)alloc((size_t)n * 2); };

  // bf16 weight copies
  unsigned short* w_sgf_ih = aU(1536LL * 300); unsigned short* w_sgf_hh = aU(1536LL * 512);
  unsigned short* w_sgb_ih = aU(1536LL * 300); unsigned short* w_sgb_hh = aU(1536LL * 512);
  unsigned short* w_cgf_ih = aU(1536LL * 300); unsigned short* w_cgf_hh = aU(1536LL * 512);
  unsigned short* w_cgb_ih = aU(1536LL * 300); unsigned short* w_cgb_hh = aU(1536LL * 512);
  unsigned short* linwb  = aU(1024LL * 6144);
  unsigned short* attnwb = aU(1024LL * 1024);
  unsigned short* c11p   = aU(3LL * 1024 * 80);    // permuted (kk,co,ci)
  unsigned short* c13p   = aU(3LL * 1024 * 1024);
  unsigned short* c21p   = aU(3LL * 1024 * 1024);
  unsigned short* c23p   = aU(3LL * 1024 * 1024);
  unsigned short* gwpb   = aU(2048LL * 1024);
  unsigned short* gwcb   = aU(2048LL * 2048);
  unsigned short* owpb   = aU(2048LL * 1024);

  // activations
  unsigned short* xsb = aU((long long)BS * En);
  unsigned short* xcb = aU(BCT * En);
  float* xg  = aF((long long)BS * G3);                // reused for every GRU pass
  float* ghb = aF((long long)BCC * G3);
  float* hf  = aF((long long)BCC * Hn);
  unsigned short* hb = aU((long long)BCC * Hn);
  float* sent = aF((long long)BS * HHn);
  unsigned short* sentb = aU((long long)BS * HHn);
  float* chf = aF((long long)BCC * Cn * HHn);
  unsigned short* chb = aU((long long)BCC * Cn * HHn);
  float* cvf = aF((long long)BCC * HHn);
  unsigned short* cvb = aU((long long)BCC * HHn);
  float* Whb   = aF((long long)BS * HHn);
  float* bhbuf = aF(BS);
  float* parsb = aF((long long)Bn * HHn);
  const int L1 = 1022, L2 = 1016, L3 = 1014, L4 = 1008;
  float* convA = aF((long long)Bn * HHn * L1);
  float* convB = aF((long long)Bn * HHn * L1);
  unsigned short* convAb = aU((long long)Bn * HHn * L1);
  unsigned short* convBb = aU((long long)Bn * HHn * L1);
  float* Pf = aF((long long)Bn * HHn);
  unsigned short* Pb = aU((long long)Bn * HHn);
  float* cs0f = aF((long long)Bn * 2048);
  unsigned short* cs0b = aU((long long)Bn * 2048);
  float* g1b = aF((long long)Bn * 2048);
  float* g2b = aF((long long)Bn * 2048);
  float* wpob = aF((long long)Bn * 2048);

  // ---- helpers -------------------------------------------------------------
  auto cvt = [&](const float* src, unsigned short* dst, long long n) {
    f32_to_bf16_knl<<<(unsigned)((n + TPB - 1) / TPB), TPB, 0, stream>>>(src, dst, n);
  };
  auto gemm = [&](int bmode, const unsigned short* A, long long lda,
                  const unsigned short* Bp, long long ldb, long long sB, int dil,
                  const float* bias, float* Cf, unsigned short* Cb,
                  long long ldc, long long sC, int M, int N, int K, int Z) {
    dim3 g((N + 63) / 64, (M + 31) / 32, Z);
    if (bmode == 0)
      gemm_bf16_wmma_t<0><<<g, dim3(TPB), 0, stream>>>(A, lda, Bp, ldb, sB, dil, bias,
                                                       Cf, Cb, ldc, sC, M, N, K);
    else if (bmode == 1)
      gemm_bf16_wmma_t<1><<<g, dim3(TPB), 0, stream>>>(A, lda, Bp, ldb, sB, dil, bias,
                                                       Cf, Cb, ldc, sC, M, N, K);
    else
      gemm_bf16_wmma_t<2><<<g, dim3(TPB), 0, stream>>>(A, lda, Bp, ldb, sB, dil, bias,
                                                       Cf, Cb, ldc, sC, M, N, K);
  };

  // ---- weight conversions --------------------------------------------------
  cvt(sgf_wih, w_sgf_ih, 1536LL * 300); cvt(sgf_whh, w_sgf_hh, 1536LL * 512);
  cvt(sgb_wih, w_sgb_ih, 1536LL * 300); cvt(sgb_whh, w_sgb_hh, 1536LL * 512);
  cvt(cgf_wih, w_cgf_ih, 1536LL * 300); cvt(cgf_whh, w_cgf_hh, 1536LL * 512);
  cvt(cgb_wih, w_cgb_ih, 1536LL * 300); cvt(cgb_whh, w_cgb_hh, 1536LL * 512);
  cvt(lin_w, linwb, 1024LL * 6144);
  cvt(attn_w, attnwb, 1024LL * 1024);
  cvt(gate_wp, gwpb, 2048LL * 1024);
  cvt(gate_wc, gwcb, 2048LL * 2048);
  cvt(out_wp, owpb, 2048LL * 1024);
  {
    long long t1 = 1024LL * 80 * 3;
    conv_w_perm_knl<<<(unsigned)((t1 + TPB - 1) / TPB), TPB, 0, stream>>>(c11_w, c11p, 1024, 80);
    long long t2 = 1024LL * 1024 * 3;
    conv_w_perm_knl<<<(unsigned)((t2 + TPB - 1) / TPB), TPB, 0, stream>>>(c13_w, c13p, 1024, 1024);
    conv_w_perm_knl<<<(unsigned)((t2 + TPB - 1) / TPB), TPB, 0, stream>>>(c21_w, c21p, 1024, 1024);
    conv_w_perm_knl<<<(unsigned)((t2 + TPB - 1) / TPB), TPB, 0, stream>>>(c23_w, c23p, 1024, 1024);
  }

  // ---- embeddings ----------------------------------------------------------
  {
    long long n1 = (long long)BS * En;
    embed_bf16_knl<<<(unsigned)((n1 + TPB - 1) / TPB), TPB, 0, stream>>>(tok_s, emb, xsb, BS, En);
    long long n2 = BCT * En;
    embed_bf16_knl<<<(unsigned)((n2 + TPB - 1) / TPB), TPB, 0, stream>>>(tok_c, emb, xcb, BCT, En);
  }

  // ---- GRU driver ----------------------------------------------------------
  auto run_gru = [&](const unsigned short* xb, int Bv, int T,
                     const unsigned short* wihb, const float* bih,
                     const unsigned short* whhb, const float* bhh,
                     float* hsF, unsigned short* hsB, int coloff, int reverse) {
    // xg = x @ w_ih^T + b_ih   (M = Bv*T, K = E, N = 1536)
    gemm(0, xb, En, wihb, En, 0, 0, bih, xg, nullptr, G3, 0, Bv * T, G3, En, 1);
    hipMemsetAsync(hf, 0, (size_t)Bv * Hn * sizeof(float), stream);
    hipMemsetAsync(hb, 0, (size_t)Bv * Hn * sizeof(unsigned short), stream);
    for (int tt = 0; tt < T; ++tt) {
      int t = reverse ? (T - 1 - tt) : tt;
      // gh = h @ w_hh^T + b_hh  (M = Bv, K = 512, N = 1536)
      gemm(0, hb, Hn, whhb, Hn, 0, 0, bhh, ghb, nullptr, G3, 0, Bv, G3, Hn, 1);
      int threads = Bv * Hn;
      gru_point_knl<<<(threads + TPB - 1) / TPB, TPB, 0, stream>>>(
          xg, ghb, hf, hb, hsF, hsB, Bv, T, t, t, coloff);
    }
  };

  run_gru(xsb, Bn, Sn, w_sgf_ih, sgf_bih, w_sgf_hh, sgf_bhh, sent, sentb, 0,  0);
  run_gru(xsb, Bn, Sn, w_sgb_ih, sgb_bih, w_sgb_hh, sgb_bhh, sent, sentb, Hn, 1);
  run_gru(xcb, BCC, Cn, w_cgf_ih, cgf_bih, w_cgf_hh, cgf_bhh, chf, chb, 0,  0);
  run_gru(xcb, BCC, Cn, w_cgb_ih, cgb_bih, w_cgb_hh, cgb_bhh, chf, chb, Hn, 1);

  // ---- choice vectors: cv = flat(ch) @ lin_w^T + lin_b ---------------------
  gemm(0, chb, 6144, linwb, 6144, 0, 0, lin_b, cvf, cvb, 1024, 0, BCC, 1024, 6144, 1);

  // ---- attention -----------------------------------------------------------
  gemm(1, sentb, HHn, attnwb, HHn, 0, 0, nullptr, Whb, nullptr, HHn, 0, BS, HHn, HHn, 1);
  bh_knl<<<(BS + TPB - 1) / TPB, TPB, 0, stream>>>(sent, attn_b, bhbuf, BS, HHn);
  attn_knl<<<Bn, TPB, 0, stream>>>(Whb, bhbuf, cvf, sent, parsb, Sn, HHn);

  // ---- conv stack (each conv = one batched WMMA GEMM; 3 TDM taps / K-chunk)
  // conv1: in sent (B, 80ch, L=1024), d=1 -> (B,1024,1022); BN1+ReLU
  gemm(2, c11p, 80, sentb, HHn, (long long)Sn * HHn, 1, nullptr,
       convA, nullptr, L1, (long long)HHn * L1, HHn, L1, 80, Bn);
  {
    long long tot = (long long)Bn * HHn * L1;
    conv_post_knl<<<(unsigned)((tot + TPB - 1) / TPB), TPB, 0, stream>>>(
        convA, convAb, c11_b, bn1_g, bn1_bb, bn1_m, bn1_v, 1, HHn, L1, tot);
  }
  // conv2: d=3 -> (B,1024,1016); bias only
  gemm(2, c13p, 1024, convAb, L1, (long long)HHn * L1, 3, nullptr,
       convB, nullptr, L2, (long long)HHn * L2, HHn, L2, 1024, Bn);
  {
    long long tot = (long long)Bn * HHn * L2;
    conv_post_knl<<<(unsigned)((tot + TPB - 1) / TPB), TPB, 0, stream>>>(
        convB, convBb, c13_b, nullptr, nullptr, nullptr, nullptr, 0, HHn, L2, tot);
  }
  // conv3: d=1 -> (B,1024,1014); BN2+ReLU
  gemm(2, c21p, 1024, convBb, L2, (long long)HHn * L2, 1, nullptr,
       convA, nullptr, L3, (long long)HHn * L3, HHn, L3, 1024, Bn);
  {
    long long tot = (long long)Bn * HHn * L3;
    conv_post_knl<<<(unsigned)((tot + TPB - 1) / TPB), TPB, 0, stream>>>(
        convA, convAb, c21_b, bn2_g, bn2_bb, bn2_m, bn2_v, 1, HHn, L3, tot);
  }
  // conv4: d=3 -> (B,1024,1008); bias only
  gemm(2, c23p, 1024, convAb, L3, (long long)HHn * L3, 3, nullptr,
       convB, nullptr, L4, (long long)HHn * L4, HHn, L4, 1024, Bn);
  {
    long long tot = (long long)Bn * HHn * L4;
    conv_post_knl<<<(unsigned)((tot + TPB - 1) / TPB), TPB, 0, stream>>>(
        convB, nullptr, c23_b, nullptr, nullptr, nullptr, nullptr, 0, HHn, L4, tot);
  }
  maxpool_knl<<<(Bn * HHn + TPB - 1) / TPB, TPB, 0, stream>>>(convB, Pf, Pb, Bn * HHn, L4);

  // ---- gated output (choice 0 only; o1 replicated x4) ----------------------
  cs0_knl<<<(Bn * 2048 + TPB - 1) / TPB, TPB, 0, stream>>>(cvf, parsb, cs0f, cs0b, Bn);
  gemm(0, Pb,   1024, gwpb, 1024, 0, 0, nullptr, g1b,  nullptr, 2048, 0, Bn, 2048, 1024, 1);
  gemm(0, cs0b, 2048, gwcb, 2048, 0, 0, nullptr, g2b,  nullptr, 2048, 0, Bn, 2048, 2048, 1);
  gemm(0, Pb,   1024, owpb, 1024, 0, 0, nullptr, wpob, nullptr, 2048, 0, Bn, 2048, 1024, 1);
  final_knl<<<Bn, TPB, 0, stream>>>(g1b, g2b, gate_b, cs0f, wpob, out_b, (float*)d_out);
}
